// RNN_16587163698034
// MI455X (gfx1250) — matrix operand order repaired and checked by reading, not compiled, not run on hardware
//
#include <hip/hip_runtime.h>
#include <hip/hip_bf16.h>
#include <math.h>

// GRU stack: 3 layers (256->512->1024->512) + Linear(512->256)+ReLU.
// bf16 WMMA (v_wmma_f32_16x16x32_bf16), fp32 accumulate & fp32 gate math.
// Batched input projections + output GEMM: register-blocked WMMA (32x64/wave).
// Time recurrence: persistent kernel with grid barrier, 16x16 tile per wave.

typedef __attribute__((ext_vector_type(16))) __bf16 v16bf;
typedef __attribute__((ext_vector_type(8)))  __bf16 v8bf;
typedef __attribute__((ext_vector_type(8)))  float  v8f;

#define BB 32
#define SS 512
#define MM (BB * SS)   // 16384 rows (b*S + s)

// ---------------------------------------------------------------- convert
__global__ void cvt_f32_bf16(const float* __restrict__ in, __bf16* __restrict__ out, int n) {
    int i = blockIdx.x * blockDim.x + threadIdx.x;
    int stride = gridDim.x * blockDim.x;
    for (; i < n; i += stride) out[i] = (__bf16)in[i];
}

// ------------------------------------------------------- WMMA fragment IO
// A fragment: 16x32 bf16 tile, row-major source with row stride K.
// lane L: row = L&15; 16 contiguous K values at offset half*16.
__device__ __forceinline__ v16bf load_a_frag(const __bf16* __restrict__ base,
                                             int K, int kk, int lane) {
    int half = lane >> 4;
    const __bf16* p = base + (size_t)(lane & 15) * K + kk + half * 16;
    v8bf lo = *(const v8bf*)p;
    v8bf hi = *(const v8bf*)(p + 8);
    return __builtin_shufflevector(lo, hi, 0,1,2,3,4,5,6,7,8,9,10,11,12,13,14,15);
}

// B fragment: 32x16 (KxN) where element (k,n) = W[n][k], W row-major [N,K].
// lane L: col n = L&15; 16 contiguous K values at offset half*16.
__device__ __forceinline__ v16bf load_b_frag(const __bf16* __restrict__ base,
                                             int K, int kk, int lane) {
    int half = lane >> 4;
    return *(const v16bf*)(base + (size_t)(lane & 15) * K + kk + half * 16);
}

// ------------------------------------------------------------ GEMM kernel
// C[M,N] = act(A_bf16[M,K] @ W_bf16[N,K]^T + bias[N]) in fp32.
// Register-blocked: each wave computes a 32x64 block (2x4 WMMA tiles),
// 8 WMMAs per 6 fragment loads per 32-deep k-step. 8 waves per block.
__global__ __launch_bounds__(256)
void wmma_gemm_bias(const __bf16* __restrict__ A, const __bf16* __restrict__ W,
                    const float* __restrict__ bias, float* __restrict__ C,
                    int M, int N, int K, int relu) {
    int wid  = (blockIdx.x * blockDim.x + threadIdx.x) >> 5;
    int lane = threadIdx.x & 31;
    int nblk = N >> 6;                 // 64-wide column blocks
    int total = (M >> 5) * nblk;       // 32-tall row blocks
    if (wid >= total) return;
    int mt = wid / nblk, nt = wid % nblk;

    const __bf16* Ab0 = A + (size_t)(mt * 32) * K;
    const __bf16* Ab1 = A + (size_t)(mt * 32 + 16) * K;
    const __bf16* Wb0 = W + (size_t)(nt * 64) * K;
    const __bf16* Wb1 = W + (size_t)(nt * 64 + 16) * K;
    const __bf16* Wb2 = W + (size_t)(nt * 64 + 32) * K;
    const __bf16* Wb3 = W + (size_t)(nt * 64 + 48) * K;

    v8f acc[2][4] = {};
    for (int kk = 0; kk < K; kk += 32) {
        __builtin_prefetch(Ab0 + (size_t)(lane & 15) * K + kk + 32, 0, 1);
        __builtin_prefetch(Ab1 + (size_t)(lane & 15) * K + kk + 32, 0, 1);
        v16bf a0 = load_a_frag(Ab0, K, kk, lane);
        v16bf a1 = load_a_frag(Ab1, K, kk, lane);
        v16bf b0 = load_b_frag(Wb0, K, kk, lane);
        v16bf b1 = load_b_frag(Wb1, K, kk, lane);
        v16bf b2 = load_b_frag(Wb2, K, kk, lane);
        v16bf b3 = load_b_frag(Wb3, K, kk, lane);
        acc[0][0] = __builtin_amdgcn_wmma_f32_16x16x32_bf16(false, a0, false, b0, (short)0, acc[0][0], false, false);
        acc[1][0] = __builtin_amdgcn_wmma_f32_16x16x32_bf16(false, a1, false, b0, (short)0, acc[1][0], false, false);
        acc[0][1] = __builtin_amdgcn_wmma_f32_16x16x32_bf16(false, a0, false, b1, (short)0, acc[0][1], false, false);
        acc[1][1] = __builtin_amdgcn_wmma_f32_16x16x32_bf16(false, a1, false, b1, (short)0, acc[1][1], false, false);
        acc[0][2] = __builtin_amdgcn_wmma_f32_16x16x32_bf16(false, a0, false, b2, (short)0, acc[0][2], false, false);
        acc[1][2] = __builtin_amdgcn_wmma_f32_16x16x32_bf16(false, a1, false, b2, (short)0, acc[1][2], false, false);
        acc[0][3] = __builtin_amdgcn_wmma_f32_16x16x32_bf16(false, a0, false, b3, (short)0, acc[0][3], false, false);
        acc[1][3] = __builtin_amdgcn_wmma_f32_16x16x32_bf16(false, a1, false, b3, (short)0, acc[1][3], false, false);
    }

    int half = lane >> 4, col = lane & 15;
#pragma unroll
    for (int mi = 0; mi < 2; ++mi) {
#pragma unroll
        for (int nj = 0; nj < 4; ++nj) {
            int ncol = nt * 64 + nj * 16 + col;
            float bv = bias[ncol];
#pragma unroll
            for (int i = 0; i < 8; ++i) {
                int mrow = mt * 32 + mi * 16 + half * 8 + i;
                float v = acc[mi][nj][i] + bv;
                if (relu) v = fmaxf(v, 0.0f);
                C[(size_t)mrow * N + ncol] = v;
            }
        }
    }
}

// ------------------------------------------------------------ grid barrier
__device__ __forceinline__ void grid_sync(unsigned* bar, unsigned& episode, unsigned nblocks) {
    __syncthreads();
    if (threadIdx.x == 0) {
        __threadfence();
        episode += nblocks;
        atomicAdd(bar, 1u);
        while (__hip_atomic_load(bar, __ATOMIC_ACQUIRE, __HIP_MEMORY_SCOPE_AGENT) < episode)
            __builtin_amdgcn_s_sleep(1);
        __threadfence();
    }
    __syncthreads();
}

// ------------------------------------------------------------ GRU recurrence
// Persistent kernel: loops over t. Phase A: hg = h @ Wh[0:2H]^T, gates r,z,
// store r*h (bf16) and z (f32). Phase C: hn = (r*h) @ Wh[2H:3H]^T,
// n = tanh(xn + hn), h = (1-z)*h + z*n. Two grid barriers per step.
__global__ __launch_bounds__(256)
void gru_scan(const float* __restrict__ xg, const __bf16* __restrict__ Wh,
              const float* __restrict__ h0, __bf16* __restrict__ ys,
              float* __restrict__ h, __bf16* __restrict__ hbf,
              __bf16* __restrict__ rhbf, float* __restrict__ zbuf,
              unsigned* bar, int H) {
    const int lane = threadIdx.x & 31;
    const int half = lane >> 4, col = lane & 15;
    const int wid = (blockIdx.x * blockDim.x + threadIdx.x) >> 5;
    const int nwaves = (gridDim.x * blockDim.x) >> 5;
    unsigned episode = 0;

    // init hidden state (h0 broadcast over batch)
    {
        int tid = blockIdx.x * blockDim.x + threadIdx.x;
        int stride = gridDim.x * blockDim.x;
        for (int i = tid; i < BB * H; i += stride) {
            float v = h0[i % H];
            h[i] = v;
            hbf[i] = (__bf16)v;
        }
    }
    grid_sync(bar, episode, gridDim.x);

    const int G3 = 3 * H;
    const int tilesA = 2 * ((2 * H) >> 4);   // M tiles (2) x N tiles (2H/16)
    const int tilesC = 2 * (H >> 4);

    for (int t = 0; t < SS; ++t) {
        // ---- Phase A: r/z gates --------------------------------------
        for (int tile = wid; tile < tilesA; tile += nwaves) {
            int mt = tile & 1, nt = tile >> 1;
            const __bf16* Abase = hbf + (size_t)mt * 16 * H;
            const __bf16* Wbase = Wh + (size_t)(nt * 16) * H;
            v8f acc = {};
            for (int kk = 0; kk < H; kk += 32) {
                v16bf a = load_a_frag(Abase, H, kk, lane);
                v16bf b = load_b_frag(Wbase, H, kk, lane);
                acc = __builtin_amdgcn_wmma_f32_16x16x32_bf16(false, a, false, b,
                                                              (short)0, acc, false, false);
            }
            int gcol = nt * 16 + col;
#pragma unroll
            for (int i = 0; i < 8; ++i) {
                int brow = mt * 16 + half * 8 + i;
                float xv = xg[((size_t)brow * SS + t) * G3 + gcol];
                float s = 1.0f / (1.0f + __expf(-(xv + acc[i])));
                if (gcol < H)
                    rhbf[brow * H + gcol] = (__bf16)(s * h[brow * H + gcol]);
                else
                    zbuf[brow * H + (gcol - H)] = s;
            }
        }
        grid_sync(bar, episode, gridDim.x);

        // ---- Phase C: candidate + state update -----------------------
        for (int tile = wid; tile < tilesC; tile += nwaves) {
            int mt = tile & 1, nt = tile >> 1;
            const __bf16* Abase = rhbf + (size_t)mt * 16 * H;
            const __bf16* Wbase = Wh + ((size_t)2 * H + nt * 16) * H;
            v8f acc = {};
            for (int kk = 0; kk < H; kk += 32) {
                v16bf a = load_a_frag(Abase, H, kk, lane);
                v16bf b = load_b_frag(Wbase, H, kk, lane);
                acc = __builtin_amdgcn_wmma_f32_16x16x32_bf16(false, a, false, b,
                                                              (short)0, acc, false, false);
            }
            int jcol = nt * 16 + col;
#pragma unroll
            for (int i = 0; i < 8; ++i) {
                int brow = mt * 16 + half * 8 + i;
                float xv = xg[((size_t)brow * SS + t) * G3 + 2 * H + jcol];
                float n = tanhf(xv + acc[i]);
                float ho = h[brow * H + jcol];
                float z = zbuf[brow * H + jcol];
                float hn = (1.0f - z) * ho + z * n;
                h[brow * H + jcol] = hn;
                __bf16 hb = (__bf16)hn;
                hbf[brow * H + jcol] = hb;
                ys[((size_t)brow * SS + t) * H + jcol] = hb;
            }
        }
        grid_sync(bar, episode, gridDim.x);
    }
}

// ---------------------------------------------------------------- launch
extern "C" void kernel_launch(void* const* d_in, const int* in_sizes, int n_in,
                              void* d_out, int out_size, void* d_ws, size_t ws_size,
                              hipStream_t stream) {
    (void)in_sizes; (void)n_in; (void)out_size; (void)ws_size;
    const float* x     = (const float*)d_in[0];
    const float* Wi[3] = {(const float*)d_in[1], (const float*)d_in[5], (const float*)d_in[9]};
    const float* Wh[3] = {(const float*)d_in[2], (const float*)d_in[6], (const float*)d_in[10]};
    const float* bh[3] = {(const float*)d_in[3], (const float*)d_in[7], (const float*)d_in[11]};
    const float* h0[3] = {(const float*)d_in[4], (const float*)d_in[8], (const float*)d_in[12]};
    const float* Wo = (const float*)d_in[13];
    const float* bo = (const float*)d_in[14];
    float* out = (float*)d_out;

    const int Is[3] = {256, 512, 1024};
    const int Hs[3] = {512, 1024, 512};

    char* base = (char*)d_ws;
    size_t off = 0;
    auto carve = [&](size_t bytes) -> char* {
        char* r = base + off;
        off = (off + bytes + 255) & ~(size_t)255;
        return r;
    };
    __bf16* bufA  = (__bf16*)carve((size_t)MM * 1024 * 2);   // ping
    __bf16* bufB  = (__bf16*)carve((size_t)MM * 1024 * 2);   // pong
    __bf16* wi_bf = (__bf16*)carve((size_t)3072 * 512 * 2);
    __bf16* wh_bf = (__bf16*)carve((size_t)3072 * 1024 * 2);
    __bf16* wo_bf = (__bf16*)carve((size_t)256 * 512 * 2);
    float*  xgbuf = (float*)carve((size_t)MM * 3072 * 4);
    float*  hbuf  = (float*)carve((size_t)32 * 1024 * 4);
    __bf16* hbf   = (__bf16*)carve((size_t)32 * 1024 * 2);
    __bf16* rhbf  = (__bf16*)carve((size_t)32 * 1024 * 2);
    float*  zbuf  = (float*)carve((size_t)32 * 1024 * 4);
    unsigned* bar = (unsigned*)carve(256);

    // x -> bf16
    cvt_f32_bf16<<<2048, 256, 0, stream>>>(x, bufA, MM * Is[0]);

    __bf16* in_b = bufA;
    __bf16* out_b = bufB;
    for (int l = 0; l < 3; ++l) {
        int I = Is[l], H = Hs[l], G = 3 * H;
        cvt_f32_bf16<<<2048, 256, 0, stream>>>(Wi[l], wi_bf, G * I);
        cvt_f32_bf16<<<2048, 256, 0, stream>>>(Wh[l], wh_bf, G * H);
        // xg = in @ Wi^T + bh  (fp32 out); 32x64 block per wave
        int waves = (MM / 32) * (G / 64);
        wmma_gemm_bias<<<waves / 8, 256, 0, stream>>>(in_b, wi_bf, bh[l], xgbuf,
                                                      MM, G, I, 0);
        hipMemsetAsync(bar, 0, sizeof(unsigned), stream);
        gru_scan<<<32, 256, 0, stream>>>(xgbuf, wh_bf, h0[l], out_b,
                                         hbuf, hbf, rhbf, zbuf, bar, H);
        __bf16* tmp = in_b; in_b = out_b; out_b = tmp;
    }

    // out = relu(ys2 @ Wo^T + bo)
    cvt_f32_bf16<<<256, 256, 0, stream>>>(Wo, wo_bf, 256 * 512);
    int waves = (MM / 32) * (256 / 64);
    wmma_gemm_bias<<<waves / 8, 256, 0, stream>>>(in_b, wo_bf, bo, out,
                                                  MM, 256, 512, 1);
}
